// DenoiseEncoderTransformer_63625645523438
// MI455X (gfx1250) — matrix operand order repaired and checked by reading, not compiled, not run on hardware
//
#include <hip/hip_runtime.h>
#include <hip/hip_bf16.h>
#include <math.h>

typedef __attribute__((ext_vector_type(16))) _Float16 v16h;
typedef __attribute__((ext_vector_type(8)))  _Float16 v8h;
typedef __attribute__((ext_vector_type(8)))  float    v8f;

#define BB 4
#define TT 2048
#define DD 128
#define VV 64
#define LL 12
#define HMM 512

static __device__ __forceinline__ v8f wmma_f16(v16h a, v16h b, v8f c) {
    return __builtin_amdgcn_wmma_f32_16x16x32_f16(false, a, false, b, (short)0, c, false, false);
}

// A-fragment (16x32 f16): lane l holds row (l&15); lanes 0-15 K={0..15},
// lanes 16-31 K={16..31}  -> one contiguous 16-half (32B) load.
static __device__ __forceinline__ v16h load_a_frag(const _Float16* p /* row base + kb */) {
    return *(const v16h*)(p);
}

// ---------------------------------------------------------------- weights fp32 -> f16
__global__ void cvt_f16_kernel(const float* __restrict__ src, _Float16* __restrict__ dst, int n) {
    int i = blockIdx.x * 256 + threadIdx.x;
    if (i < n) dst[i] = (_Float16)src[i];
}

// ---------------------------------------------------------------- embedding + pos
__global__ void embed_kernel(const int* __restrict__ x, const float* __restrict__ pos,
                             float* __restrict__ H) {
    int i  = blockIdx.x * 256 + threadIdx.x;      // over B*T*D
    int d  = i & (DD - 1);
    int bt = i >> 7;                              // b*T + t
    int t  = bt & (TT - 1);
    float e = 0.0f;
    if (d < VV) { float diff = (float)d - (float)x[bt]; e = -0.5f * diff * diff; }
    H[i] = e + pos[t * DD + d];
}

// ---------------------------------------------------------------- layernorm -> f16
__global__ void ln_kernel(const float* __restrict__ H, const float* __restrict__ g,
                          const float* __restrict__ be, _Float16* __restrict__ out, int rows) {
    int lane = threadIdx.x & 31;
    int wv   = threadIdx.x >> 5;
    int row  = blockIdx.x * 4 + wv;
    if (row >= rows) return;                      // wave-uniform
    const float* h = H + (size_t)row * DD;
    float v[4];
    float s = 0.0f;
#pragma unroll
    for (int i = 0; i < 4; ++i) { v[i] = h[lane + 32 * i]; s += v[i]; }
#pragma unroll
    for (int m = 1; m <= 16; m <<= 1) s += __shfl_xor(s, m, 32);
    float mu  = s * (1.0f / DD);
    float var = 0.0f;
#pragma unroll
    for (int i = 0; i < 4; ++i) { float d = v[i] - mu; var += d * d; }
#pragma unroll
    for (int m = 1; m <= 16; m <<= 1) var += __shfl_xor(var, m, 32);
    float rstd = rsqrtf(var * (1.0f / DD) + 1e-5f);
#pragma unroll
    for (int i = 0; i < 4; ++i) {
        int d = lane + 32 * i;
        out[(size_t)row * DD + d] = (_Float16)((v[i] - mu) * rstd * g[d] + be[d]);
    }
}

// ---------------------------------------------------------------- WMMA GEMM (16x64 per wave)
// C[m,n] = act( sum_k A[m,k] * Bw[n,k] + bias[n] )
// Each wave owns a 16x64 output strip: 4 accumulators reuse the A fragment 4x.
// outH: optional f16 output ([M,N] or transposed-per-batch [b,N,Tdim])
// outR: optional fp32 residual accumulate (+=) into [M,N]
__global__ void gemm_wmma_kernel(const _Float16* __restrict__ A, const _Float16* __restrict__ Bw,
                                 const float* __restrict__ bias, _Float16* __restrict__ outH,
                                 float* __restrict__ outR, int M, int N, int K,
                                 int doGelu, int transT, int Tdim) {
    int lane    = threadIdx.x & 31;
    int wv      = threadIdx.x >> 5;
    int gw      = blockIdx.x * 8 + wv;
    int stripsN = N >> 6;                          // 64-wide strips
    int tilesM  = M >> 4;
    if (gw >= tilesM * stripsN) return;            // wave-uniform
    int tM = gw / stripsN;
    int sN = gw % stripsN;

    int rowA = tM * 16 + (lane & 15);
    int colB = sN * 64 + (lane & 15);
    int kbA  = (lane < 16) ? 0 : 16;
    int kbB  = (lane < 16) ? 0 : 16;
    const _Float16* pa = A  + (size_t)rowA * K + kbA;
    const _Float16* pb = Bw + (size_t)colB * K + kbB;
    const size_t    nStride = (size_t)16 * K;      // 16 columns ahead in weight rows

    v8f c0 = {}, c1 = {}, c2 = {}, c3 = {};
    for (int k0 = 0; k0 < K; k0 += 32) {
        __builtin_prefetch(pa + k0 + 128, 0, 0);
        __builtin_prefetch(pb + k0 + 128, 0, 0);
        v16h a  = load_a_frag(pa + k0);
        v16h b0 = *(const v16h*)(pb + k0);
        v16h b1 = *(const v16h*)(pb + nStride + k0);
        v16h b2 = *(const v16h*)(pb + 2 * nStride + k0);
        v16h b3 = *(const v16h*)(pb + 3 * nStride + k0);
        c0 = wmma_f16(a, b0, c0);
        c1 = wmma_f16(a, b1, c1);
        c2 = wmma_f16(a, b2, c2);
        c3 = wmma_f16(a, b3, c3);
    }

    int half = lane >> 4;
    v8f acc[4] = {c0, c1, c2, c3};
#pragma unroll
    for (int nn = 0; nn < 4; ++nn) {
        int n = sN * 64 + nn * 16 + (lane & 15);
        float bval = bias ? bias[n] : 0.0f;
#pragma unroll
        for (int r = 0; r < 8; ++r) {
            int   m   = tM * 16 + r + 8 * half;
            float val = acc[nn][r] + bval;
            if (doGelu) val = 0.5f * val * (1.0f + erff(val * 0.70710678118654752f));
            if (outR) outR[(size_t)m * N + n] += val;
            if (outH) {
                if (transT) {
                    int bb = m / Tdim, t = m % Tdim;
                    outH[((size_t)bb * N + n) * Tdim + t] = (_Float16)val;
                } else {
                    outH[(size_t)m * N + n] = (_Float16)val;
                }
            }
        }
    }
}

// ---------------------------------------------------------------- flash attention
// One wave per 16 query rows. q,k: [B*T, D] f16.  vt: [B, D, T] f16 (per-batch transposed).
// H (fp32 residual stream) gets += attn @ v.
__global__ void flash_attn_kernel(const _Float16* __restrict__ q, const _Float16* __restrict__ k,
                                  const _Float16* __restrict__ vt, float* __restrict__ H) {
    __shared__ __align__(32) _Float16 ldsP[8][16 * 32];
    const float rs = 0.08838834764831845f;        // 1/sqrt(128)

    int lane = threadIdx.x & 31;
    int wv   = threadIdx.x >> 5;
    int gw   = blockIdx.x * 8 + wv;               // 0..511
    int qt   = gw & (TT / 16 - 1);
    int b    = gw >> 7;
    int row  = lane & 15;
    int half = lane >> 4;
    int kbA  = half ? 16 : 0;
    int kbB  = half ? 16 : 0;

    // Q fragments for this wave's 16 rows (K = D = 128 -> 4 chunks of 32)
    const _Float16* qrow = q + ((size_t)(b * TT) + qt * 16 + row) * DD;
    v16h aq[4];
#pragma unroll
    for (int c = 0; c < 4; ++c) aq[c] = load_a_frag(qrow + c * 32 + kbA);

    v8f  o[8];
    float mrow[8], lrow[8];
#pragma unroll
    for (int t = 0; t < 8; ++t) o[t] = (v8f){};
#pragma unroll
    for (int r = 0; r < 8; ++r) { mrow[r] = -1e30f; lrow[r] = 0.0f; }

    for (int j = 0; j < TT; j += 32) {
        // S = Q @ K^T for 16 x 32 keys (two 16x16 tiles)
        v8f s0 = {}, s1 = {};
#pragma unroll
        for (int c = 0; c < 4; ++c) {
            v16h b0 = *(const v16h*)(k + ((size_t)(b * TT) + j +      (lane & 15)) * DD + c * 32 + kbB);
            v16h b1 = *(const v16h*)(k + ((size_t)(b * TT) + j + 16 + (lane & 15)) * DD + c * 32 + kbB);
            s0 = wmma_f16(aq[c], b0, s0);
            s1 = wmma_f16(aq[c], b1, s1);
        }
        // online softmax per row; stage P into per-wave LDS as f16
#pragma unroll
        for (int r = 0; r < 8; ++r) {
            float x0 = s0[r] * rs, x1 = s1[r] * rs;
            float tm = fmaxf(x0, x1);
#pragma unroll
            for (int m = 1; m <= 8; m <<= 1) tm = fmaxf(tm, __shfl_xor(tm, m, 32));
            float mn    = fmaxf(mrow[r], tm);
            float alpha = __expf(mrow[r] - mn);
            float p0 = __expf(x0 - mn), p1 = __expf(x1 - mn);
            float rsum = p0 + p1;
#pragma unroll
            for (int m = 1; m <= 8; m <<= 1) rsum += __shfl_xor(rsum, m, 32);
            lrow[r] = lrow[r] * alpha + rsum;
            mrow[r] = mn;
#pragma unroll
            for (int t = 0; t < 8; ++t) o[t][r] *= alpha;
            int prow = r + 8 * half;
            ldsP[wv][prow * 32 +      (lane & 15)] = (_Float16)p0;
            ldsP[wv][prow * 32 + 16 + (lane & 15)] = (_Float16)p1;
        }
        asm volatile("s_wait_dscnt 0" ::: "memory");   // order LDS write -> read (CDNA5 split counters)
        v16h ap = load_a_frag(&ldsP[wv][row * 32 + kbA]);
        // O += P @ V   (V^T rows are contiguous over keys)
#pragma unroll
        for (int t = 0; t < 8; ++t) {
            int d = t * 16 + (lane & 15);
            v16h bv = *(const v16h*)(vt + ((size_t)(b * DD) + d) * TT + j + kbB);
            o[t] = wmma_f16(ap, bv, o[t]);
        }
    }
    // epilogue: divide by softmax denominator, residual add into H
#pragma unroll
    for (int r = 0; r < 8; ++r) {
        float inv = 1.0f / lrow[r];
        int   m   = b * TT + qt * 16 + r + 8 * half;
#pragma unroll
        for (int t = 0; t < 8; ++t) {
            int d = t * 16 + (lane & 15);
            H[(size_t)m * DD + d] += o[t][r] * inv;
        }
    }
}

// ---------------------------------------------------------------- readout
__global__ void readout_kernel(const float* __restrict__ H, const float* __restrict__ Wro,
                               const float* __restrict__ bro, float* __restrict__ out) {
    int m = blockIdx.x * 256 + threadIdx.x;       // 0..B*T-1
    const float* h = H + (size_t)m * DD;
    float s = bro[0];
#pragma unroll 8
    for (int d = 0; d < DD; ++d) s += h[d] * Wro[d];
    out[m] = s;
}

// ================================================================ host
extern "C" void kernel_launch(void* const* d_in, const int* in_sizes, int n_in,
                              void* d_out, int out_size, void* d_ws, size_t ws_size,
                              hipStream_t stream) {
    const int*   x    = (const int*)  d_in[0];
    const float* pos  = (const float*)d_in[1];
    const float* Wq   = (const float*)d_in[2];
    const float* Wk   = (const float*)d_in[3];
    const float* Wv   = (const float*)d_in[4];
    const float* ln1g = (const float*)d_in[5];
    const float* ln1b = (const float*)d_in[6];
    const float* W1   = (const float*)d_in[7];
    const float* b1   = (const float*)d_in[8];
    const float* W2   = (const float*)d_in[9];
    const float* b2   = (const float*)d_in[10];
    const float* ln2g = (const float*)d_in[11];
    const float* ln2b = (const float*)d_in[12];
    const float* Wro  = (const float*)d_in[13];
    const float* bro  = (const float*)d_in[14];
    float* out = (float*)d_out;

    const int M = BB * TT;                         // 8192 rows
    char*  base = (char*)d_ws;
    size_t off  = 0;
    auto alloc = [&](size_t bytes) -> char* {
        char* p = base + off;
        off += (bytes + 255) & ~(size_t)255;
        return p;
    };
    float*    H     = (float*)   alloc((size_t)M * DD * 4);
    _Float16* h16   = (_Float16*)alloc((size_t)M * DD * 2);      // LN1/LN2 output
    _Float16* q16   = (_Float16*)alloc((size_t)M * DD * 2);
    _Float16* k16   = (_Float16*)alloc((size_t)M * DD * 2);
    _Float16* vt16  = (_Float16*)alloc((size_t)M * DD * 2);      // [B, D, T]
    _Float16* mlp1  = (_Float16*)alloc((size_t)M * HMM * 2);
    _Float16* Wq16  = (_Float16*)alloc((size_t)LL * DD * DD * 2);
    _Float16* Wk16  = (_Float16*)alloc((size_t)LL * DD * DD * 2);
    _Float16* Wv16  = (_Float16*)alloc((size_t)LL * DD * DD * 2);
    _Float16* W116  = (_Float16*)alloc((size_t)LL * HMM * DD * 2);
    _Float16* W216  = (_Float16*)alloc((size_t)LL * DD * HMM * 2);

    // weights -> f16
    {
        int n = LL * DD * DD;
        cvt_f16_kernel<<<(n + 255) / 256, 256, 0, stream>>>(Wq, Wq16, n);
        cvt_f16_kernel<<<(n + 255) / 256, 256, 0, stream>>>(Wk, Wk16, n);
        cvt_f16_kernel<<<(n + 255) / 256, 256, 0, stream>>>(Wv, Wv16, n);
        int n2 = LL * HMM * DD;
        cvt_f16_kernel<<<(n2 + 255) / 256, 256, 0, stream>>>(W1, W116, n2);
        cvt_f16_kernel<<<(n2 + 255) / 256, 256, 0, stream>>>(W2, W216, n2);
    }
    // embedding
    embed_kernel<<<(M * DD) / 256, 256, 0, stream>>>(x, pos, H);

    const int lnBlocks    = (M + 3) / 4;
    const int qkvBlocks   = ((M / 16) * (DD / 64) + 7) / 8;       // 128
    const int mlp1Blocks  = ((M / 16) * (HMM / 64) + 7) / 8;      // 512
    const int flashBlocks = (BB * (TT / 16)) / 8;                 // 64

    for (int l = 0; l < LL; ++l) {
        // ---- attention block
        ln_kernel<<<lnBlocks, 128, 0, stream>>>(H, ln1g + l * DD, ln1b + l * DD, h16, M);
        gemm_wmma_kernel<<<qkvBlocks, 256, 0, stream>>>(h16, Wq16 + (size_t)l * DD * DD, nullptr,
                                                        q16, nullptr, M, DD, DD, 0, 0, TT);
        gemm_wmma_kernel<<<qkvBlocks, 256, 0, stream>>>(h16, Wk16 + (size_t)l * DD * DD, nullptr,
                                                        k16, nullptr, M, DD, DD, 0, 0, TT);
        gemm_wmma_kernel<<<qkvBlocks, 256, 0, stream>>>(h16, Wv16 + (size_t)l * DD * DD, nullptr,
                                                        vt16, nullptr, M, DD, DD, 0, 1, TT);
        flash_attn_kernel<<<flashBlocks, 256, 0, stream>>>(q16, k16, vt16, H);
        // ---- MLP block
        ln_kernel<<<lnBlocks, 128, 0, stream>>>(H, ln2g + l * DD, ln2b + l * DD, h16, M);
        gemm_wmma_kernel<<<mlp1Blocks, 256, 0, stream>>>(h16, W116 + (size_t)l * HMM * DD, b1 + l * HMM,
                                                         mlp1, nullptr, M, HMM, DD, 1, 0, TT);
        gemm_wmma_kernel<<<qkvBlocks, 256, 0, stream>>>(mlp1, W216 + (size_t)l * DD * HMM, b2 + l * DD,
                                                        nullptr, H, M, DD, HMM, 0, 0, TT);
    }
    readout_kernel<<<M / 256, 256, 0, stream>>>(H, Wro, bro, out);
    (void)in_sizes; (void)n_in; (void)out_size; (void)ws_size;
}
